// HSTGL_25640954757833
// MI455X (gfx1250) — matrix operand order repaired and checked by reading, not compiled, not run on hardware
//
#include <hip/hip_runtime.h>
#include <math.h>

// ---------------------------------------------------------------------------
// MAGNN metapath GRU + edge-softmax aggregation for MI455X (gfx1250).
// GEMMs (373 GFLOP) run on v_wmma_f32_16x16x32_bf16; hidden state is
// LDS-resident bf16, weights are L2-resident bf16.
// ---------------------------------------------------------------------------

#define HID   512
#define G3    1536     // 3*HID
#define DD    64       // OUT_DIM
#define NH    8        // heads

typedef __bf16 v16bf __attribute__((ext_vector_type(16)));
typedef float  v8f   __attribute__((ext_vector_type(8)));

union Frag { v16bf v; unsigned u[8]; };

__device__ __forceinline__ v8f zero8() {
    v8f z = {0.f,0.f,0.f,0.f,0.f,0.f,0.f,0.f};
    return z;
}

__device__ __forceinline__ unsigned short f32_bf16(float f) {
    unsigned u = __float_as_uint(f);
    unsigned r = u + 0x7FFFu + ((u >> 16) & 1u);   // RNE
    return (unsigned short)(r >> 16);
}
__device__ __forceinline__ float bf16_f32(unsigned short s) {
    return __uint_as_float(((unsigned)s) << 16);
}

__device__ __forceinline__ v8f wmma_bf16(const Frag& a, const Frag& b, v8f c) {
    return __builtin_amdgcn_wmma_f32_16x16x32_bf16(
        /*neg_a=*/false, a.v, /*neg_b=*/false, b.v,
        /*c_mod=*/(short)0, c, /*reuse_a=*/false, /*reuse_b=*/false);
}

// B fragment: 32x16 bf16, row-major weight matrix W[row..row+15][k0..k0+31].
// lanes 0-15 : N = lane,    K = 0..7, 16..23
// lanes 16-31: N = lane-16, K = 8..15, 24..31
__device__ __forceinline__ Frag loadB(const unsigned short* __restrict__ W,
                                      int ld, int row, int k0, int ln, int g) {
    Frag b;
    const unsigned* p = (const unsigned*)(W + (size_t)(row + ln) * ld + k0 + 8 * g);
#pragma unroll
    for (int q = 0; q < 8; ++q) b.u[q] = p[(q < 4) ? q : (q + 4)];
    return b;
}

// A fragment from global bf16 rows (stride 64): 16x32, M = lane&15.
// VGPR v<4: K = 2v + 8g (+k0); v>=4: K = 16 + 2(v-4) + 8g (+k0).
__device__ __forceinline__ Frag loadAx(const unsigned short* __restrict__ X,
                                       int row, int k0, int g) {
    Frag a;
    const unsigned short* p = X + (size_t)row * DD;
#pragma unroll
    for (int v = 0; v < 8; ++v) {
        int K = k0 + ((v < 4) ? (2 * v + 8 * g) : (16 + 2 * (v - 4) + 8 * g));
        a.u[v] = *(const unsigned*)(p + K);
    }
    return a;
}

__device__ __forceinline__ float sigmoidf_(float x) {
    return 1.f / (1.f + __expf(-x));
}

// ---------------------------------------------------------------------------
// Kernel 1: convert weights f32 -> bf16 (W_ih: 1536x64, W_hh: 1536x512)
// ---------------------------------------------------------------------------
__global__ void hstgl_cvt_weights(const float* __restrict__ Wih,
                                  const float* __restrict__ Whh,
                                  unsigned short* __restrict__ Wih_bf,
                                  unsigned short* __restrict__ Whh_bf) {
    int i = blockIdx.x * blockDim.x + threadIdx.x;
    const int NIH = G3 * DD;          // 98304
    const int NHH = G3 * HID;         // 786432
    if (i < NIH) Wih_bf[i] = f32_bf16(Wih[i]);
    else if (i < NIH + NHH) Whh_bf[i - NIH] = f32_bf16(Whh[i - NIH]);
}

// ---------------------------------------------------------------------------
// Kernel 2: gather metapath features -> bf16  Xg[t][e][d]
// ---------------------------------------------------------------------------
__global__ void hstgl_gather(const float* __restrict__ features,
                             const int* __restrict__ mp_idx,   // [E][3]
                             unsigned short* __restrict__ Xg,  // [3][E][64]
                             int E) {
    int i = blockIdx.x * blockDim.x + threadIdx.x;
    int total = 3 * E * DD;
    if (i >= total) return;
    int d = i & (DD - 1);
    int r = i / DD;          // t*E + e
    int t = r / E;
    int e = r - t * E;
    int node = mp_idx[e * 3 + t];
    Xg[i] = f32_bf16(features[(size_t)node * DD + d]);
}

// ---------------------------------------------------------------------------
// Kernel 3: init output / segment buffers
// ---------------------------------------------------------------------------
__global__ void hstgl_init(float* __restrict__ out, float* __restrict__ amax,
                           float* __restrict__ denom, int out_n, int seg_n) {
    int i = blockIdx.x * blockDim.x + threadIdx.x;
    if (i < out_n) out[i] = 0.f;
    if (i < seg_n) {
        amax[i]  = __uint_as_float(0xFF800000u);   // -inf
        denom[i] = 0.f;
    }
}

// ---------------------------------------------------------------------------
// Kernel 4: fused 3-step GRU over metapaths via WMMA bf16.
// One block = 32 edges, 8 waves; wave owns 4 column-tiles of 16 (HID=512).
// ---------------------------------------------------------------------------
__global__ void __launch_bounds__(256)
hstgl_gru(const unsigned short* __restrict__ Xg,     // [3][E][64] bf16
          const unsigned short* __restrict__ Wih,    // [1536][64]  bf16
          const unsigned short* __restrict__ Whh,    // [1536][512] bf16
          const float* __restrict__ b_ih,
          const float* __restrict__ b_hh,
          float* __restrict__ eft,                   // [E][512] f32
          int E) {
    __shared__ unsigned short hbuf[2][32][HID];      // 64 KB, bf16 ping-pong

    const int lane = threadIdx.x & 31;
    const int wave = threadIdx.x >> 5;
    const int g    = lane >> 4;
    const int ln   = lane & 15;
    const int e0   = blockIdx.x * 32;

    for (int t = 0; t < 3; ++t) {
        const int cur = t & 1, nxt = cur ^ 1;
        const unsigned short* Xt = Xg + (size_t)t * E * DD;

        for (int jj = 0; jj < 4; ++jj) {
            const int jt = wave + 8 * jj;            // column tile 0..31
            const int jR = jt * 16;
            const int jZ = HID + jt * 16;
            const int jN = 2 * HID + jt * 16;

            v8f accR[2], accZ[2], accNi[2], accNh[2];
#pragma unroll
            for (int mi = 0; mi < 2; ++mi) {
                accR[mi] = zero8(); accZ[mi] = zero8();
                accNi[mi] = zero8(); accNh[mi] = zero8();
            }

            // ---- x_t @ W_ih^T  (K = 64) ----
#pragma unroll
            for (int k0 = 0; k0 < DD; k0 += 32) {
                Frag bR = loadB(Wih, DD, jR, k0, ln, g);
                Frag bZ = loadB(Wih, DD, jZ, k0, ln, g);
                Frag bN = loadB(Wih, DD, jN, k0, ln, g);
#pragma unroll
                for (int mi = 0; mi < 2; ++mi) {
                    Frag a = loadAx(Xt, e0 + mi * 16 + ln, k0, g);
                    accR[mi]  = wmma_bf16(a, bR, accR[mi]);
                    accZ[mi]  = wmma_bf16(a, bZ, accZ[mi]);
                    accNi[mi] = wmma_bf16(a, bN, accNi[mi]);
                }
            }

            // ---- h @ W_hh^T (K = 512); skipped at t=0 since h0 == 0 ----
            if (t > 0) {
                for (int k0 = 0; k0 < HID; k0 += 32) {
                    Frag bR = loadB(Whh, HID, jR, k0, ln, g);
                    Frag bZ = loadB(Whh, HID, jZ, k0, ln, g);
                    Frag bN = loadB(Whh, HID, jN, k0, ln, g);
#pragma unroll
                    for (int mi = 0; mi < 2; ++mi) {
                        Frag a;
#pragma unroll
                        for (int v = 0; v < 8; ++v) {
                            int K = k0 + ((v < 4) ? (2 * v + 8 * g)
                                                  : (16 + 2 * (v - 4) + 8 * g));
                            a.u[v] = *(const unsigned*)&hbuf[cur][mi * 16 + ln][K];
                        }
                        accR[mi]  = wmma_bf16(a, bR, accR[mi]);
                        accZ[mi]  = wmma_bf16(a, bZ, accZ[mi]);
                        accNh[mi] = wmma_bf16(a, bN, accNh[mi]);
                    }
                }
            }

            // ---- gates + h update, entirely in registers ----
            // C layout: VGPR v holds row (v + 8g), col = jt*16 + ln
            const int j = jt * 16 + ln;
            const float bri = b_ih[j],           brh = b_hh[j];
            const float bzi = b_ih[HID + j],     bzh = b_hh[HID + j];
            const float bni = b_ih[2 * HID + j], bnh = b_hh[2 * HID + j];

#pragma unroll
            for (int mi = 0; mi < 2; ++mi) {
#pragma unroll
                for (int v = 0; v < 8; ++v) {
                    const int m = mi * 16 + v + 8 * g;
                    float r = sigmoidf_(accR[mi][v] + bri + brh);
                    float z = sigmoidf_(accZ[mi][v] + bzi + bzh);
                    float n = tanhf(accNi[mi][v] + bni + r * (accNh[mi][v] + bnh));
                    float hprev = (t > 0) ? bf16_f32(hbuf[cur][m][j]) : 0.f;
                    float hnew  = (1.f - z) * n + z * hprev;
                    if (t == 2) {
                        if (e0 + m < E)
                            eft[(size_t)(e0 + m) * HID + j] = hnew;
                    } else {
                        hbuf[nxt][m][j] = f32_bf16(hnew);
                    }
                }
            }
        }
        __syncthreads();   // hbuf[nxt] complete before next step reads it
    }
}

// ---------------------------------------------------------------------------
// Kernel 5: attention logits + leaky relu + segment max (atomic)
// ---------------------------------------------------------------------------
__device__ __forceinline__ void atomicMaxFloat(float* addr, float v) {
    if (v >= 0.f) atomicMax((int*)addr, __float_as_int(v));
    else          atomicMin((unsigned int*)addr, __float_as_uint(v));
}

__global__ void hstgl_logits(const float* __restrict__ eft,
                             const float* __restrict__ attn,
                             const int* __restrict__ dst,
                             float* __restrict__ aL,
                             float* __restrict__ amax, int E) {
    int i = blockIdx.x * blockDim.x + threadIdx.x;
    if (i >= E * NH) return;
    int e = i >> 3, h = i & 7;
    const float* pe = eft + (size_t)e * HID + h * DD;
    const float* pa = attn + h * DD;
    float s = 0.f;
#pragma unroll
    for (int d = 0; d < DD; ++d) s += pe[d] * pa[d];
    s = (s >= 0.f) ? s : 0.01f * s;          // leaky relu
    aL[i] = s;
    atomicMaxFloat(&amax[dst[e] * NH + h], s);
}

// ---------------------------------------------------------------------------
// Kernel 6: ea = exp(a - amax[dst]); denom = segment_sum(ea)
// ---------------------------------------------------------------------------
__global__ void hstgl_expsum(const int* __restrict__ dst,
                             float* __restrict__ aL,     // in: a, out: ea
                             const float* __restrict__ amax,
                             float* __restrict__ denom, int E) {
    int i = blockIdx.x * blockDim.x + threadIdx.x;
    if (i >= E * NH) return;
    int e = i >> 3, h = i & 7;
    int d0 = dst[e];
    float v = __expf(aL[i] - amax[d0 * NH + h]);
    aL[i] = v;
    unsafeAtomicAdd(&denom[d0 * NH + h], v);
}

// ---------------------------------------------------------------------------
// Kernel 7: weighted message scatter-sum into out[N][8][64]
// ---------------------------------------------------------------------------
__global__ void hstgl_aggregate(const float* __restrict__ eft,
                                const float* __restrict__ aL,    // ea
                                const float* __restrict__ denom,
                                const int* __restrict__ dst,
                                float* __restrict__ out, int E) {
    int e = blockIdx.x;
    if (e >= E) return;
    int d0 = dst[e];
    for (int idx = threadIdx.x; idx < HID; idx += blockDim.x) {
        int h = idx >> 6;
        float alpha = aL[e * NH + h] / denom[d0 * NH + h];
        unsafeAtomicAdd(&out[(size_t)d0 * HID + idx],
                        eft[(size_t)e * HID + idx] * alpha);
    }
}

// ---------------------------------------------------------------------------
// Host launcher
// ---------------------------------------------------------------------------
extern "C" void kernel_launch(void* const* d_in, const int* in_sizes, int n_in,
                              void* d_out, int out_size, void* d_ws, size_t ws_size,
                              hipStream_t stream) {
    const float* features = (const float*)d_in[0];
    const float* W_ih     = (const float*)d_in[1];
    const float* W_hh     = (const float*)d_in[2];
    const float* b_ih     = (const float*)d_in[3];
    const float* b_hh     = (const float*)d_in[4];
    const float* attn     = (const float*)d_in[5];
    const int*   mp_idx   = (const int*)d_in[6];
    const int*   edge_dst = (const int*)d_in[7];

    const int E = in_sizes[7];
    const int N = in_sizes[0] / DD;

    // workspace carve-out (256B aligned)
    char* ws = (char*)d_ws;
    size_t off = 0;
    auto take = [&](size_t bytes) -> char* {
        char* p = ws + off;
        off = (off + bytes + 255) & ~(size_t)255;
        return p;
    };
    unsigned short* Wih_bf = (unsigned short*)take((size_t)G3 * DD * 2);
    unsigned short* Whh_bf = (unsigned short*)take((size_t)G3 * HID * 2);
    unsigned short* Xg     = (unsigned short*)take((size_t)3 * E * DD * 2);
    float* eft   = (float*)take((size_t)E * HID * 4);
    float* aL    = (float*)take((size_t)E * NH * 4);
    float* amax  = (float*)take((size_t)N * NH * 4);
    float* denom = (float*)take((size_t)N * NH * 4);
    (void)ws_size; (void)n_in;

    float* out = (float*)d_out;

    // 1) weights -> bf16
    {
        int total = G3 * DD + G3 * HID;
        hstgl_cvt_weights<<<(total + 255) / 256, 256, 0, stream>>>(
            W_ih, W_hh, Wih_bf, Whh_bf);
    }
    // 2) gather metapath features -> bf16
    {
        int total = 3 * E * DD;
        hstgl_gather<<<(total + 255) / 256, 256, 0, stream>>>(
            features, mp_idx, Xg, E);
    }
    // 3) init out / amax / denom
    {
        int total = out_size;  // N*512 >= N*8
        hstgl_init<<<(total + 255) / 256, 256, 0, stream>>>(
            out, amax, denom, out_size, N * NH);
    }
    // 4) fused GRU (WMMA bf16)
    {
        int blocks = (E + 31) / 32;
        hstgl_gru<<<blocks, 256, 0, stream>>>(
            Xg, Wih_bf, Whh_bf, b_ih, b_hh, eft, E);
    }
    // 5) attention logits + segment max
    hstgl_logits<<<(E * NH + 255) / 256, 256, 0, stream>>>(
        eft, attn, edge_dst, aL, amax, E);
    // 6) exp + segment sum
    hstgl_expsum<<<(E * NH + 255) / 256, 256, 0, stream>>>(
        edge_dst, aL, amax, denom, E);
    // 7) weighted scatter aggregation
    hstgl_aggregate<<<E, 256, 0, stream>>>(
        eft, aL, denom, edge_dst, out, E);
}
